// MultiHeadAttention_90374701843033
// MI455X (gfx1250) — matrix operand order repaired and checked by reading, not compiled, not run on hardware
//
#include <hip/hip_runtime.h>

// ---------------------------------------------------------------------------
// MHA block for B=2, L=2048, D=512, H=8, DK=DV=64 on gfx1250 (wave32, WMMA).
// ---------------------------------------------------------------------------

typedef __attribute__((ext_vector_type(16))) _Float16 v16h;
typedef __attribute__((ext_vector_type(8)))  _Float16 v8h;
typedef __attribute__((ext_vector_type(4)))  _Float16 v4h;
typedef __attribute__((ext_vector_type(8)))  float    v8f;
typedef __attribute__((ext_vector_type(4)))  float    v4f;
typedef __attribute__((ext_vector_type(4)))  unsigned char v4uc;

union V16 { v16h v; v8h h[2]; };

#define L_SEQ 2048
#define D_MODEL 512
#define N_HEAD 8
#define D_HEAD 64
#define BATCH 2
#define LPAD 2052           // 2048 + 4 pad floats -> kills LDS bank conflicts

static __device__ __forceinline__ v8f wmma_f16(v16h a, v16h b, v8f c) {
    return __builtin_amdgcn_wmma_f32_16x16x32_f16(false, a, false, b, (short)0, c,
                                                  false, false);
}

// ---------------------------------------------------------------------------
// f32 -> f16 elementwise convert (4 elems / thread)
// ---------------------------------------------------------------------------
__global__ void cvt_f16_kernel(const float* __restrict__ in,
                               _Float16* __restrict__ out, int n4) {
    int i = blockIdx.x * 256 + threadIdx.x;
    if (i < n4) {
        v4f f = ((const v4f*)in)[i];
        v4h o;
        o[0] = (_Float16)f[0]; o[1] = (_Float16)f[1];
        o[2] = (_Float16)f[2]; o[3] = (_Float16)f[3];
        ((v4h*)out)[i] = o;
    }
}

// ---------------------------------------------------------------------------
// GEMM: C[M=4096, N=512] = A[4096,512] @ W[512,512]^T (+bias [+resid])
// One wave -> one 16x16 tile.  MODE 0: write qh/kh [B,H,L,64] f16
//                              MODE 1: write vh^T  [B,H,64,L] f16
//                              MODE 2: write f32 proj + bias + residual
// ---------------------------------------------------------------------------
template <int MODE>
__global__ __launch_bounds__(256) void gemm_kernel(
    const _Float16* __restrict__ A, const _Float16* __restrict__ W,
    const float* __restrict__ bias, const float* __restrict__ resid,
    void* __restrict__ outp) {
    const int tid  = threadIdx.x;
    const int wave = tid >> 5, lane = tid & 31;
    const int g = lane >> 4, nl = lane & 15;

    const int tile = blockIdx.x * 8 + wave;     // 8192 tiles total
    const int tm = tile >> 5;                   // 256 row tiles
    const int tn = tile & 31;                   // 32  col tiles

    const _Float16* Arow = A + (size_t)(tm * 16 + nl) * D_MODEL;
    const _Float16* Wrow = W + (size_t)(tn * 16 + nl) * D_MODEL;

    v8f acc = {};
#pragma unroll
    for (int kk = 0; kk < D_MODEL; kk += 32) {
        V16 a;
        a.h[0] = *(const v8h*)(Arow + kk + 16 * g);
        a.h[1] = *(const v8h*)(Arow + kk + 8 + 16 * g);
        v16h bb = *(const v16h*)(Wrow + kk + 16 * g);
        acc = wmma_f16(a.v, bb, acc);
    }

    const int n  = tn * 16 + nl;
    const float bn = bias[n];
#pragma unroll
    for (int r = 0; r < 8; ++r) {
        const int mrow = tm * 16 + r + 8 * g;   // global row in [0,4096)
        float val = acc[r] + bn;
        if (MODE == 0) {                        // qh/kh: [B,H,L,64] f16
            const int b = mrow >> 11, l = mrow & (L_SEQ - 1);
            const int h = n >> 6, dk = n & 63;
            ((_Float16*)outp)[(((size_t)(b * N_HEAD + h) * L_SEQ) + l) * D_HEAD + dk] =
                (_Float16)val;
        } else if (MODE == 1) {                 // vh^T: [B,H,64,L] f16
            const int b = mrow >> 11, l = mrow & (L_SEQ - 1);
            const int h = n >> 6, dv = n & 63;
            ((_Float16*)outp)[(((size_t)(b * N_HEAD + h) * D_HEAD) + dv) * L_SEQ + l] =
                (_Float16)val;
        } else {                                // out proj f32 + residual
            val += resid[(size_t)mrow * D_MODEL + n];
            ((float*)outp)[(size_t)mrow * D_MODEL + n] = val;
        }
    }
}

// ---------------------------------------------------------------------------
// Fused attention: block = (b, h, 16 query rows).  8 waves x 256 key cols.
// S kept in LDS; mask applied in the (coalesced) row-max scan; softmax in
// LDS; normalized attn streamed write-only to d_out with NON-TEMPORAL b128
// stores (256 MB stream > 192 MB L2 -> do not pollute the cache);
// ctx = softmax(S) @ V via WMMA with cross-wave LDS reduction.
// ---------------------------------------------------------------------------
__global__ __launch_bounds__(256) void attn_kernel(
    const _Float16* __restrict__ qh, const _Float16* __restrict__ kh,
    const _Float16* __restrict__ vt, const unsigned char* __restrict__ mask,
    float* __restrict__ attn_out, _Float16* __restrict__ ctx_out) {
    __shared__ float sS[16 * LPAD];
    __shared__ float sMax[16];
    __shared__ float sInv[16];

    const int tid  = threadIdx.x;
    const int wave = tid >> 5, lane = tid & 31;
    const int g = lane >> 4, nl = lane & 15;

    const int blk = blockIdx.x;          // b*H*128 + h*128 + qt
    const int qt = blk & 127;
    const int bh = blk >> 7;             // b*8 + h
    const int b = bh >> 3, h = bh & 7;
    const int lq0 = qt * 16;

    const _Float16* Q = qh + ((size_t)bh * L_SEQ + lq0) * D_HEAD;
    const _Float16* K = kh + (size_t)bh * L_SEQ * D_HEAD;
    const _Float16* V = vt + (size_t)bh * D_HEAD * L_SEQ;

    // Q tile A-operands for the two 32-wide K steps (kept in registers)
    V16 aQ0, aQ1;
    {
        const _Float16* qrow = Q + (size_t)nl * D_HEAD;
        aQ0.h[0] = *(const v8h*)(qrow + 16 * g);
        aQ0.h[1] = *(const v8h*)(qrow + 8 + 16 * g);
        aQ1.h[0] = *(const v8h*)(qrow + 32 + 16 * g);
        aQ1.h[1] = *(const v8h*)(qrow + 40 + 16 * g);
    }

    // ---- scores: each wave fills 256 columns of S[16][2048] ----
    for (int c = 0; c < 16; ++c) {
        const int lk0 = (wave * 16 + c) * 16;
        v8f acc = {};
        const _Float16* krow = K + (size_t)(lk0 + nl) * D_HEAD + 16 * g;
        v16h b0 = *(const v16h*)(krow);
        v16h b1 = *(const v16h*)(krow + 32);
        acc = wmma_f16(aQ0.v, b0, acc);
        acc = wmma_f16(aQ1.v, b1, acc);
#pragma unroll
        for (int r = 0; r < 8; ++r) {
            const int m = r + 8 * g;
            sS[m * LPAD + lk0 + nl] = acc[r];
        }
    }
    __syncthreads();

    // ---- mask + row max: thread -> (row r, float4 groups), coalesced ----
    {
        const int r = tid >> 4, c0 = tid & 15;
        const unsigned char* mrowp = mask + ((size_t)b * L_SEQ + lq0 + r) * L_SEQ;
        const float NEG = -__builtin_inff();
        float mx = NEG;
        for (int i = 0; i < 32; ++i) {
            const int c4 = (c0 + 16 * i) * 4;          // 4-wide column group
            v4f vv = *(const v4f*)&sS[r * LPAD + c4];
            const v4uc mv =
                __builtin_nontemporal_load((const v4uc*)(mrowp + c4));
            if (mv[0] | mv[1] | mv[2] | mv[3]) {
                if (mv[0]) vv[0] = NEG;
                if (mv[1]) vv[1] = NEG;
                if (mv[2]) vv[2] = NEG;
                if (mv[3]) vv[3] = NEG;
                *(v4f*)&sS[r * LPAD + c4] = vv;
            }
            mx = fmaxf(mx, fmaxf(fmaxf(vv[0], vv[1]), fmaxf(vv[2], vv[3])));
        }
#pragma unroll
        for (int m = 8; m >= 1; m >>= 1) mx = fmaxf(mx, __shfl_xor(mx, m, 16));
        if (c0 == 0) sMax[r] = mx;
    }
    __syncthreads();

    // ---- exp + row sum (exp stored back to LDS, unnormalized) ----
    {
        const int r = tid >> 4, c0 = tid & 15;
        const float mx = sMax[r];
        float s = 0.f;
        for (int i = 0; i < 32; ++i) {
            const int c4 = (c0 + 16 * i) * 4;
            v4f vv = *(const v4f*)&sS[r * LPAD + c4];
            vv[0] = __expf(vv[0] - mx);
            vv[1] = __expf(vv[1] - mx);
            vv[2] = __expf(vv[2] - mx);
            vv[3] = __expf(vv[3] - mx);
            *(v4f*)&sS[r * LPAD + c4] = vv;
            s += vv[0] + vv[1] + vv[2] + vv[3];
        }
#pragma unroll
        for (int m = 8; m >= 1; m >>= 1) s += __shfl_xor(s, m, 16);
        if (c0 == 0) sInv[r] = 1.0f / s;
    }
    __syncthreads();

    // ---- stream normalized attn to global: non-temporal b128 stores ----
    // attn_flat layout: [h, b, lq, lk]
    float* arow = attn_out + (((size_t)h * BATCH + b) * L_SEQ + lq0) * L_SEQ;
    for (int r = 0; r < 16; ++r) {
        const float inv = sInv[r];
        const v4f* src = (const v4f*)&sS[r * LPAD];
        v4f* dst = (v4f*)(arow + (size_t)r * L_SEQ);
#pragma unroll
        for (int j = 0; j < 2; ++j) {
            const int c = tid + 256 * j;               // 512 float4 per row
            v4f t = src[c];
            t *= inv;
            __builtin_nontemporal_store(t, dst + c);
        }
    }

    // ---- ctx partials: wave handles its own 256 key columns ----
    v8f acc0 = {}, acc1 = {}, acc2 = {}, acc3 = {};
    const int cbase = wave * 256;
#pragma unroll 1
    for (int s = 0; s < 8; ++s) {
        const int ck = cbase + 32 * s;
        const v4f f0 = *(const v4f*)&sS[nl * LPAD + ck + 16 * g];
        const v4f f1 = *(const v4f*)&sS[nl * LPAD + ck + 16 * g + 4];
        const v4f f2 = *(const v4f*)&sS[nl * LPAD + ck + 16 * g + 8];
        const v4f f3 = *(const v4f*)&sS[nl * LPAD + ck + 16 * g + 12];
        V16 a;
        a.v[0]  = (_Float16)f0[0]; a.v[1]  = (_Float16)f0[1];
        a.v[2]  = (_Float16)f0[2]; a.v[3]  = (_Float16)f0[3];
        a.v[4]  = (_Float16)f1[0]; a.v[5]  = (_Float16)f1[1];
        a.v[6]  = (_Float16)f1[2]; a.v[7]  = (_Float16)f1[3];
        a.v[8]  = (_Float16)f2[0]; a.v[9]  = (_Float16)f2[1];
        a.v[10] = (_Float16)f2[2]; a.v[11] = (_Float16)f2[3];
        a.v[12] = (_Float16)f3[0]; a.v[13] = (_Float16)f3[1];
        a.v[14] = (_Float16)f3[2]; a.v[15] = (_Float16)f3[3];
        const _Float16* vb = V + (size_t)nl * L_SEQ + ck + 16 * g;
        v16h b0 = *(const v16h*)(vb);
        v16h b1 = *(const v16h*)(vb + 16 * L_SEQ);
        v16h b2 = *(const v16h*)(vb + 32 * L_SEQ);
        v16h b3 = *(const v16h*)(vb + 48 * L_SEQ);
        acc0 = wmma_f16(a.v, b0, acc0);
        acc1 = wmma_f16(a.v, b1, acc1);
        acc2 = wmma_f16(a.v, b2, acc2);
        acc3 = wmma_f16(a.v, b3, acc3);
    }
    __syncthreads();                 // everyone done reading sS
    float* sCtx = sS;                // reuse LDS for 8 x [16x64] partials
#pragma unroll
    for (int r = 0; r < 8; ++r) {
        const int m = r + 8 * g;
        sCtx[wave * 1024 + m * 64 + 0  + nl] = acc0[r];
        sCtx[wave * 1024 + m * 64 + 16 + nl] = acc1[r];
        sCtx[wave * 1024 + m * 64 + 32 + nl] = acc2[r];
        sCtx[wave * 1024 + m * 64 + 48 + nl] = acc3[r];
    }
    __syncthreads();

    // ---- cross-wave reduce, scale by 1/rowsum, write ctx f16 [B,L,H*DV] ----
    for (int e = tid; e < 1024; e += 256) {
        float s = 0.f;
#pragma unroll
        for (int w = 0; w < 8; ++w) s += sCtx[w * 1024 + e];
        const int m = e >> 6, dv = e & 63;
        s *= sInv[m];
        ctx_out[((size_t)(b * L_SEQ) + lq0 + m) * D_MODEL + h * D_HEAD + dv] =
            (_Float16)s;
    }
}

// ---------------------------------------------------------------------------
// LayerNorm: one wave per 512-elem row (32 lanes x 16 elems)
// ---------------------------------------------------------------------------
__global__ __launch_bounds__(256) void ln_kernel(
    const float* __restrict__ x, const float* __restrict__ gamma,
    const float* __restrict__ beta, float* __restrict__ out) {
    const int wave = threadIdx.x >> 5, lane = threadIdx.x & 31;
    const int row = blockIdx.x * 8 + wave;
    const float* xr = x + (size_t)row * D_MODEL;

    v4f v[4];
    float s = 0.f, sq = 0.f;
#pragma unroll
    for (int i = 0; i < 4; ++i) {
        v[i] = *(const v4f*)(xr + lane * 16 + i * 4);
        s  += v[i][0] + v[i][1] + v[i][2] + v[i][3];
        sq += v[i][0] * v[i][0] + v[i][1] * v[i][1] + v[i][2] * v[i][2] +
              v[i][3] * v[i][3];
    }
#pragma unroll
    for (int m = 16; m >= 1; m >>= 1) {
        s  += __shfl_xor(s, m, 32);
        sq += __shfl_xor(sq, m, 32);
    }
    const float mu  = s * (1.0f / D_MODEL);
    const float var = sq * (1.0f / D_MODEL) - mu * mu;
    const float inv = rsqrtf(var + 1e-5f);

    float* orow = out + (size_t)row * D_MODEL;
#pragma unroll
    for (int i = 0; i < 4; ++i) {
        const int c = lane * 16 + i * 4;
        const v4f gv = *(const v4f*)(gamma + c);
        const v4f bv = *(const v4f*)(beta + c);
        v4f o;
        o[0] = (v[i][0] - mu) * inv * gv[0] + bv[0];
        o[1] = (v[i][1] - mu) * inv * gv[1] + bv[1];
        o[2] = (v[i][2] - mu) * inv * gv[2] + bv[2];
        o[3] = (v[i][3] - mu) * inv * gv[3] + bv[3];
        *(v4f*)(orow + c) = o;
    }
}

// ---------------------------------------------------------------------------
// Host launcher
// ---------------------------------------------------------------------------
extern "C" void kernel_launch(void* const* d_in, const int* in_sizes, int n_in,
                              void* d_out, int out_size, void* d_ws, size_t ws_size,
                              hipStream_t stream) {
    (void)in_sizes; (void)n_in; (void)out_size; (void)ws_size;

    const float* q    = (const float*)d_in[0];
    const float* k    = (const float*)d_in[1];
    const float* v    = (const float*)d_in[2];
    const unsigned char* mask = (const unsigned char*)d_in[3];
    const float* Wq   = (const float*)d_in[4];
    const float* bq   = (const float*)d_in[5];
    const float* Wk   = (const float*)d_in[6];
    const float* bk   = (const float*)d_in[7];
    const float* Wv   = (const float*)d_in[8];
    const float* bv   = (const float*)d_in[9];
    const float* Wo   = (const float*)d_in[10];
    const float* bo   = (const float*)d_in[11];
    const float* gamma = (const float*)d_in[12];
    const float* beta  = (const float*)d_in[13];

    const size_t ML   = (size_t)BATCH * L_SEQ;        // 4096
    const size_t nQKV = ML * D_MODEL;                 // 2,097,152
    const size_t nW   = (size_t)D_MODEL * D_MODEL;    // 262,144

    char* ws = (char*)d_ws;
    size_t off = 0;
    _Float16* q16  = (_Float16*)(ws + off); off += nQKV * 2;
    _Float16* k16  = (_Float16*)(ws + off); off += nQKV * 2;
    _Float16* v16  = (_Float16*)(ws + off); off += nQKV * 2;
    _Float16* Wq16 = (_Float16*)(ws + off); off += nW * 2;
    _Float16* Wk16 = (_Float16*)(ws + off); off += nW * 2;
    _Float16* Wv16 = (_Float16*)(ws + off); off += nW * 2;
    _Float16* Wo16 = (_Float16*)(ws + off); off += nW * 2;
    _Float16* qh   = (_Float16*)(ws + off); off += nQKV * 2;   // [B,H,L,64]
    _Float16* kh   = (_Float16*)(ws + off); off += nQKV * 2;   // [B,H,L,64]
    _Float16* vt   = (_Float16*)(ws + off); off += nQKV * 2;   // [B,H,64,L]
    _Float16* ctx16 = (_Float16*)(ws + off); off += nQKV * 2;  // [B,L,512]
    float*    tmp  = (float*)(ws + off);    off += nQKV * 4;   // proj f32

    float* out_f = (float*)d_out;                       // [B,L,512]
    float* attn_f = out_f + nQKV;                       // [H*B, L, L]

    // 1) f32 -> f16 converts (4 elems / thread)
    const int g4q = (int)((nQKV / 4 + 255) / 256);
    const int g4w = (int)((nW / 4 + 255) / 256);
    cvt_f16_kernel<<<g4q, 256, 0, stream>>>(q, q16, (int)(nQKV / 4));
    cvt_f16_kernel<<<g4q, 256, 0, stream>>>(k, k16, (int)(nQKV / 4));
    cvt_f16_kernel<<<g4q, 256, 0, stream>>>(v, v16, (int)(nQKV / 4));
    cvt_f16_kernel<<<g4w, 256, 0, stream>>>(Wq, Wq16, (int)(nW / 4));
    cvt_f16_kernel<<<g4w, 256, 0, stream>>>(Wk, Wk16, (int)(nW / 4));
    cvt_f16_kernel<<<g4w, 256, 0, stream>>>(Wv, Wv16, (int)(nW / 4));
    cvt_f16_kernel<<<g4w, 256, 0, stream>>>(Wo, Wo16, (int)(nW / 4));

    // 2) QKV projections (8192 16x16 tiles -> 1024 blocks of 8 waves)
    gemm_kernel<0><<<1024, 256, 0, stream>>>(q16, Wq16, bq, nullptr, qh);
    gemm_kernel<0><<<1024, 256, 0, stream>>>(k16, Wk16, bk, nullptr, kh);
    gemm_kernel<1><<<1024, 256, 0, stream>>>(v16, Wv16, bv, nullptr, vt);

    // 3) fused attention: B*H*(L/16) = 2048 blocks
    attn_kernel<<<BATCH * N_HEAD * (L_SEQ / 16), 256, 0, stream>>>(
        qh, kh, vt, mask, attn_f, ctx16);

    // 4) output projection + bias + residual (f32)
    gemm_kernel<2><<<1024, 256, 0, stream>>>(ctx16, Wo16, bo, q, tmp);

    // 5) LayerNorm -> final output
    ln_kernel<<<(int)(ML / 8), 256, 0, stream>>>(tmp, gamma, beta, out_f);
}
